// Megalodon_42202348650930
// MI455X (gfx1250) — matrix operand order repaired and checked by reading, not compiled, not run on hardware
//
#include <hip/hip_runtime.h>
#include <hip/hip_bf16.h>
#include <cstdint>
#include <cstddef>

#define HIDDEN   256
#define HEADS    8
#define DHEAD    32
#define NGRAPH   32
#define NODESPER 48
#define NNODE    (NGRAPH * NODESPER)          // 1536
#define EPERG    (NODESPER * (NODESPER - 1))  // 2256
#define NEDGE    (NGRAPH * EPERG)             // 72192
#define DISTD    128
#define FEIN     (3 * HIDDEN + DISTD)         // 896
#define SW       1024

static_assert(NNODE % 64 == 0, "M tiling");
static_assert(NEDGE % 64 == 0, "M tiling");

typedef __attribute__((ext_vector_type(16))) __bf16 vbf16;
typedef __attribute__((ext_vector_type(8)))  float  v8f;

enum { A_PLAIN = 0, A_SILU = 1, A_BF16 = 2, A_FEATS = 3, A_SUMPAIR = 4, A_CATMD = 5 };
enum { EP_NONE = 0, EP_BIAS = 1, EP_BIAS_SILU = 2, EP_GATED = 3 };

struct GArgs {
  const float* A;  int lda;
  const float* W;  int ldw;
  const float* bias;
  float*       out; int ldo;
  int M, N, K;
  const int*   src; const int* tgt;
  const float* xn;  const float* en; const float* dist;
  const float* res; int ldres;
  const float* gate; int ldg; int goff;
};

__device__ __forceinline__ float fast_sigmoid(float x) {
  return __builtin_amdgcn_rcpf(1.f + __expf(-x));
}

__device__ __forceinline__ unsigned pack2bf(float a, float b) {
  union { __bf16 h[2]; unsigned u; } p;
  p.h[0] = (__bf16)a; p.h[1] = (__bf16)b;
  return p.u;
}

// Load 4 consecutive-K fp32 values of op(A) at (row, k); k is a multiple of 4 and
// every concat-segment boundary is a multiple of 4, so float4 loads are safe.
template<int AM>
__device__ __forceinline__ void loadA4(const GArgs& g, int row, int k, float v[4]) {
  float4 f;
  if (AM == A_PLAIN || AM == A_SILU) {
    f = *(const float4*)&g.A[(size_t)row * g.lda + k];
  } else if (AM == A_BF16) {
    const unsigned* p = (const unsigned*)((const __bf16*)g.A + (size_t)row * g.lda + k);
    unsigned a = p[0], b = p[1];
    f.x = __uint_as_float(a << 16); f.y = __uint_as_float(a & 0xffff0000u);
    f.z = __uint_as_float(b << 16); f.w = __uint_as_float(b & 0xffff0000u);
  } else if (AM == A_FEATS) {
    if (k < 256)      f = *(const float4*)&g.xn[(size_t)g.src[row] * 256 + k];
    else if (k < 512) f = *(const float4*)&g.xn[(size_t)g.tgt[row] * 256 + (k - 256)];
    else if (k < 768) f = *(const float4*)&g.en[(size_t)row * 256 + (k - 512)];
    else              f = *(const float4*)&g.dist[(size_t)row * 128 + (k - 768)];
  } else if (AM == A_SUMPAIR) {
    float4 a = *(const float4*)&g.A[(size_t)g.src[row] * 256 + k];
    float4 b = *(const float4*)&g.A[(size_t)g.tgt[row] * 256 + k];
    f.x = a.x + b.x; f.y = a.y + b.y; f.z = a.z + b.z; f.w = a.w + b.w;
  } else { // A_CATMD
    f = (k < 256) ? *(const float4*)&g.A[(size_t)row * 256 + k]
                  : *(const float4*)&g.dist[(size_t)row * 128 + (k - 256)];
  }
  v[0] = f.x; v[1] = f.y; v[2] = f.z; v[3] = f.w;
  if (AM == A_SILU) {
    #pragma unroll
    for (int j = 0; j < 4; j++) v[j] = v[j] * fast_sigmoid(v[j]);
  }
}

// Store 4 consecutive-K values for row r into fragment-major A LDS (packed 8B store).
// Fragment layout (ISA 16x32 bf16 A): lane = (r&15) + 16*((k>>3)&1), elem i = (k>>4)*8 + (k&7).
__device__ __forceinline__ void storeAfrag(__bf16 (*Af)[32][16], int r, int k, const float v[4]) {
  int half = (k >> 3) & 1;
  int i0   = ((k >> 4) << 3) + (k & 7);
  uint2 u2; u2.x = pack2bf(v[0], v[1]); u2.y = pack2bf(v[2], v[3]);
  *(uint2*)&Af[r >> 4][(r & 15) + (half << 4)][i0] = u2;
}

// Stage one 32x32 B tile (fragment-major).  (k,n): lane = (n&15)+16*((k>>3)&1), elem = (k>>4)*8 + (k&7).
__device__ __forceinline__ void stageBtile(__bf16 (*Bf)[32][16], const float* W, size_t ldw,
                                           int k0, int colBase, int tid) {
  int r  = tid >> 3;             // k row 0..31
  int c4 = (tid & 7) * 4;        // n col
  float4 w = *(const float4*)&W[(size_t)(k0 + r) * ldw + colBase + c4];
  int sub = c4 >> 4;
  int lb  = (c4 & 15) + (((r >> 3) & 1) << 4);
  int ii  = ((r >> 4) << 3) + (r & 7);
  Bf[sub][lb + 0][ii] = (__bf16)w.x;
  Bf[sub][lb + 1][ii] = (__bf16)w.y;
  Bf[sub][lb + 2][ii] = (__bf16)w.z;
  Bf[sub][lb + 3][ii] = (__bf16)w.w;
}

// ---------------------------------------------------------------------------
// Generic WMMA bf16 GEMM: out[M,N] = epi( op(A)[M,K] @ W[K,N] )
// Block: 256 thr = 8 waves -> 64 x (32*NT) output tile; waves 4(M) x 2(N),
// each wave owns NT accumulators.  A tile staged once per K-chunk and reused
// by NT independent WMMAs (cuts A-stream traffic NT x; feeds XDL pipeline with
// independent accumulation chains).  LDS tiles are in WMMA fragment order so
// each lane loads its v16bf fragment as one aligned 32B load (2x ds_load_b128).
// ---------------------------------------------------------------------------
template<int AM, int EP, int NT>
__global__ __launch_bounds__(256) void gemm_wmma(GArgs g) {
  __shared__ alignas(32) __bf16 Afrag[4][32][16];       // [m-subtile][lane][elem]
  __shared__ alignas(32) __bf16 Bfrag[NT][2][32][16];   // [ntile][n-subtile][lane][elem]
  const int tid  = threadIdx.x;
  const int lane = tid & 31;
  const int wave = tid >> 5;
  const int wm   = wave >> 1;   // 0..3
  const int wn   = wave & 1;    // 0..1
  const int rowBase = blockIdx.x * 64;
  const int colBase = blockIdx.y * (32 * NT);
  const int half = lane >> 4;
  const int l15  = lane & 15;

  v8f acc[NT];
  #pragma unroll
  for (int nt = 0; nt < NT; nt++) acc[nt] = (v8f){0.f,0.f,0.f,0.f,0.f,0.f,0.f,0.f};

  for (int k0 = 0; k0 < g.K; k0 += 32) {
    // A tile: 64 rows x 32 k, 4 elems/thread/iter, 2 iters.
    #pragma unroll
    for (int it = 0; it < 2; it++) {
      int i = tid + it * 256;        // 0..511 quads
      int r = i >> 3;
      int k = (i & 7) * 4;
      float v[4];
      loadA4<AM>(g, rowBase + r, k0 + k, v);
      storeAfrag(Afrag, r, k, v);
    }
    // B tiles: NT x (32 k x 32 n), one float4 per thread per tile.
    #pragma unroll
    for (int nt = 0; nt < NT; nt++)
      stageBtile(Bfrag[nt], g.W, (size_t)g.ldw, k0, colBase + nt * 32, tid);
    // Prefetch next K chunk (weights stream + plain activation stream).
    if (k0 + 32 < g.K) {
      #pragma unroll
      for (int nt = 0; nt < NT; nt++)
        __builtin_prefetch(&g.W[(size_t)(k0 + 32 + (tid >> 3)) * g.ldw +
                                colBase + nt * 32 + (tid & 7) * 4], 0, 1);
      if (AM == A_PLAIN || AM == A_SILU || AM == A_BF16)
        __builtin_prefetch(&g.A[(size_t)(rowBase + (tid & 63)) * g.lda + k0 + 32], 0, 1);
    }
    __syncthreads();
    vbf16 af = *(const vbf16*)(&Afrag[wm][lane][0]);
    #pragma unroll
    for (int nt = 0; nt < NT; nt++) {
      vbf16 bfv = *(const vbf16*)(&Bfrag[nt][wn][lane][0]);
      acc[nt] = __builtin_amdgcn_wmma_f32_16x16x32_bf16(false, af, false, bfv,
                                                        (short)0, acc[nt], false, false);
    }
    __syncthreads();
  }

  #pragma unroll
  for (int nt = 0; nt < NT; nt++) {
    const int col = colBase + nt * 32 + wn * 16 + l15;
    #pragma unroll
    for (int r = 0; r < 8; r++) {
      int row = rowBase + wm * 16 + half * 8 + r;           // ISA C/D layout
      float v = acc[nt][r];
      if (EP == EP_BIAS)      v += g.bias[col];
      if (EP == EP_BIAS_SILU) { v += g.bias[col]; v = v * fast_sigmoid(v); }
      if (EP == EP_GATED) {
        float gt = g.gate[(size_t)row * g.ldg + g.goff + col];
        v = g.res[(size_t)row * g.ldres + col] + gt * v;
      }
      g.out[(size_t)row * g.ldo + col] = v;
    }
  }
}

// ---------------------------------------------------------------------------
// Fused SwiGLU in-projection: out[M,SW](bf16) = silu(A@W[:, :SW]) * (A@W[:, SW:])
// NT=4 column tiles per block (128 of the SW columns), dual-B per tile.
// ---------------------------------------------------------------------------
#define SGNT 4
__global__ __launch_bounds__(256) void gemm_swiglu(const float* A, int lda,
                                                   const float* W, int ldw,
                                                   __bf16* out, int ldo, int Mrows) {
  __shared__ alignas(32) __bf16 Afrag[4][32][16];
  __shared__ alignas(32) __bf16 Ba[SGNT][2][32][16];
  __shared__ alignas(32) __bf16 Bb[SGNT][2][32][16];
  const int tid  = threadIdx.x;
  const int lane = tid & 31;
  const int wave = tid >> 5;
  const int wm = wave >> 1, wn = wave & 1;
  const int rowBase = blockIdx.x * 64;
  const int colBase = blockIdx.y * (32 * SGNT);
  const int half = lane >> 4, l15 = lane & 15;
  (void)Mrows;

  v8f accA[SGNT], accB[SGNT];
  #pragma unroll
  for (int nt = 0; nt < SGNT; nt++) {
    accA[nt] = (v8f){0.f,0.f,0.f,0.f,0.f,0.f,0.f,0.f};
    accB[nt] = (v8f){0.f,0.f,0.f,0.f,0.f,0.f,0.f,0.f};
  }

  for (int k0 = 0; k0 < HIDDEN; k0 += 32) {
    #pragma unroll
    for (int it = 0; it < 2; it++) {
      int i = tid + it * 256;
      int r = i >> 3;
      int k = (i & 7) * 4;
      float4 f = *(const float4*)&A[(size_t)(rowBase + r) * lda + k0 + k];
      float v[4] = {f.x, f.y, f.z, f.w};
      storeAfrag(Afrag, r, k, v);
    }
    #pragma unroll
    for (int nt = 0; nt < SGNT; nt++) {
      stageBtile(Ba[nt], W, (size_t)ldw, k0, colBase + nt * 32, tid);
      stageBtile(Bb[nt], W, (size_t)ldw, k0, SW + colBase + nt * 32, tid);
    }
    if (k0 + 32 < HIDDEN) {
      #pragma unroll
      for (int nt = 0; nt < SGNT; nt++)
        __builtin_prefetch(&W[(size_t)(k0 + 32 + (tid >> 3)) * ldw +
                              colBase + nt * 32 + (tid & 7) * 4], 0, 1);
      __builtin_prefetch(&A[(size_t)(rowBase + (tid & 63)) * lda + k0 + 32], 0, 1);
    }
    __syncthreads();
    vbf16 af = *(const vbf16*)(&Afrag[wm][lane][0]);
    #pragma unroll
    for (int nt = 0; nt < SGNT; nt++) {
      vbf16 b1 = *(const vbf16*)(&Ba[nt][wn][lane][0]);
      vbf16 b2 = *(const vbf16*)(&Bb[nt][wn][lane][0]);
      accA[nt] = __builtin_amdgcn_wmma_f32_16x16x32_bf16(false, af, false, b1, (short)0, accA[nt], false, false);
      accB[nt] = __builtin_amdgcn_wmma_f32_16x16x32_bf16(false, af, false, b2, (short)0, accB[nt], false, false);
    }
    __syncthreads();
  }
  #pragma unroll
  for (int nt = 0; nt < SGNT; nt++) {
    const int col = colBase + nt * 32 + wn * 16 + l15;
    #pragma unroll
    for (int r = 0; r < 8; r++) {
      int row = rowBase + wm * 16 + half * 8 + r;
      float a = accA[nt][r], b = accB[nt][r];
      out[(size_t)row * ldo + col] = (__bf16)(a * fast_sigmoid(a) * b);
    }
  }
}

// ---------------------------------------------------------------------------
// Per-graph LayerNorm stats: mean/rstd over rpg rows x 256 feats (regular batching)
// ---------------------------------------------------------------------------
__global__ __launch_bounds__(256) void bln_stats(const float* X, int ldx, int coff,
                                                 int rpg, float eps,
                                                 float* mean, float* rstd) {
  int gph = blockIdx.x, tid = threadIdx.x;
  float s = 0.f, s2 = 0.f;
  int totq = rpg * (HIDDEN / 4);
  size_t rbase = (size_t)gph * rpg;
  for (int i = tid; i < totq; i += 256) {
    int r = i >> 6, c = (i & 63) * 4;
    float4 v = *(const float4*)&X[(rbase + r) * (size_t)ldx + coff + c];
    s  += v.x + v.y + v.z + v.w;
    s2 += v.x * v.x + v.y * v.y + v.z * v.z + v.w * v.w;
  }
  __shared__ float sh[256], sh2[256];
  sh[tid] = s; sh2[tid] = s2;
  __syncthreads();
  for (int o = 128; o > 0; o >>= 1) {
    if (tid < o) { sh[tid] += sh[tid + o]; sh2[tid] += sh2[tid + o]; }
    __syncthreads();
  }
  if (tid == 0) {
    float cnt = (float)(rpg * HIDDEN);
    float m = sh[0] / cnt;
    float var = fmaxf(sh2[0] / cnt - m * m, 0.f);
    mean[gph] = m;
    rstd[gph] = rsqrtf(var + eps);
  }
}

template<bool MOD, bool WB>
__global__ __launch_bounds__(256) void bln_apply(const float* X, int ldx, int coff,
                                                 float* out, int ldo, int rpg,
                                                 const float* mean, const float* rstd,
                                                 const float* mods, int ldm, int shOff, int scOff,
                                                 const float* w, const float* b, int rows) {
  int idx = blockIdx.x * 256 + threadIdx.x;
  if (idx >= rows * 256) return;
  int row = idx >> 8, c = idx & 255;
  int gph = row / rpg;
  float v = (X[(size_t)row * ldx + coff + c] - mean[gph]) * rstd[gph];
  if (MOD) {
    size_t mb = (size_t)row * ldm;
    v = v * (1.f + mods[mb + (size_t)scOff + c]) + mods[mb + (size_t)shOff + c];
  }
  if (WB) v = v * w[c] + b[c];
  out[(size_t)row * ldo + c] = v;
}

// scatter_mean over src: edges of node n are exactly [n*47, (n+1)*47)
__global__ __launch_bounds__(256) void agg_mean(const float* msg, float* xagg) {
  int idx = blockIdx.x * 256 + threadIdx.x;
  if (idx >= NNODE * HIDDEN) return;
  int n = idx >> 8, c = idx & 255;
  float s = 0.f;
  size_t base = (size_t)n * 47 * 256 + c;
  for (int j = 0; j < 47; j++) s += msg[base + (size_t)j * 256];
  xagg[idx] = s * (1.f / 47.f);
}

// Block-diagonal attention: one block per (graph, head); 48-node sequence.
__global__ __launch_bounds__(64) void attn_kernel(const float* QN, const float* KN,
                                                  const float* QKV, float* ATTN) {
  int gph = blockIdx.x, h = blockIdx.y;
  __shared__ float ks[NODESPER][DHEAD];
  __shared__ float vs[NODESPER][DHEAD];
  int tid = threadIdx.x;
  for (int i = tid; i < NODESPER * DHEAD; i += 64) {
    int j = i >> 5, d = i & 31;
    ks[j][d] = KN[(size_t)(gph * NODESPER + j) * HIDDEN + h * DHEAD + d];
    vs[j][d] = QKV[(size_t)(gph * NODESPER + j) * 768 + 512 + h * DHEAD + d];
  }
  __syncthreads();
  if (tid < NODESPER) {
    float q[DHEAD];
    for (int d = 0; d < DHEAD; d++)
      q[d] = QN[(size_t)(gph * NODESPER + tid) * HIDDEN + h * DHEAD + d];
    const float scale = 0.17677669529663687f;  // 1/sqrt(32)
    float sc[NODESPER];
    float mx = -1e30f;
    for (int j = 0; j < NODESPER; j++) {
      float s = 0.f;
      for (int d = 0; d < DHEAD; d++) s += q[d] * ks[j][d];
      s *= scale; sc[j] = s; mx = fmaxf(mx, s);
    }
    float sum = 0.f;
    for (int j = 0; j < NODESPER; j++) { float e = __expf(sc[j] - mx); sc[j] = e; sum += e; }
    float inv = __builtin_amdgcn_rcpf(sum);
    float o[DHEAD];
    for (int d = 0; d < DHEAD; d++) o[d] = 0.f;
    for (int j = 0; j < NODESPER; j++) {
      float wj = sc[j] * inv;
      for (int d = 0; d < DHEAD; d++) o[d] += wj * vs[j][d];
    }
    for (int d = 0; d < DHEAD; d++)
      ATTN[(size_t)(gph * NODESPER + tid) * HIDDEN + h * DHEAD + d] = o[d];
  }
}

// ---------------------------------------------------------------------------
extern "C" void kernel_launch(void* const* d_in, const int* in_sizes, int n_in,
                              void* d_out, int out_size, void* d_ws, size_t ws_size,
                              hipStream_t stream) {
  (void)in_sizes; (void)n_in; (void)out_size; (void)ws_size;
  // params (dict order), then batch, x, t_emb_h, edge_attr, edge_index, t_emb_e, dist, edge_batch
  const float* adaln_w      = (const float*)d_in[0];
  const float* adaln_b      = (const float*)d_in[1];
  const float* adaln_e_w    = (const float*)d_in[2];
  const float* adaln_e_b    = (const float*)d_in[3];
  const float* fe_w0        = (const float*)d_in[4];
  const float* fe_b0        = (const float*)d_in[5];
  const float* fe_w1        = (const float*)d_in[6];
  const float* fe_b1        = (const float*)d_in[7];
  const float* qkv_w        = (const float*)d_in[8];
  const float* out_w        = (const float*)d_in[9];
  const float* ffn_w_in     = (const float*)d_in[10];
  const float* ffn_w_out    = (const float*)d_in[11];
  const float* ffn_norm_w   = (const float*)d_in[12];
  const float* ffn_norm_b   = (const float*)d_in[13];
  const float* lin_e0_w     = (const float*)d_in[14];
  const float* lin_e1_w     = (const float*)d_in[15];
  const float* ffn_e_w_in   = (const float*)d_in[16];
  const float* ffn_e_w_out  = (const float*)d_in[17];
  const float* ffn_norm_e_w = (const float*)d_in[18];
  const float* ffn_norm_e_b = (const float*)d_in[19];
  const float* xin          = (const float*)d_in[21];
  const float* t_emb_h      = (const float*)d_in[22];
  const float* edge_attr    = (const float*)d_in[23];
  const int*   edge_index   = (const int*)d_in[24];
  const float* t_emb_e      = (const float*)d_in[25];
  const float* dist         = (const float*)d_in[26];
  const int* src = edge_index;
  const int* tgt = edge_index + NEDGE;

  float* out_x = (float*)d_out;
  float* out_e = out_x + (size_t)NNODE * HIDDEN;

  float* ws = (float*)d_ws;
  size_t off = 0;
  auto alloc = [&](size_t n) { float* p = ws + off; off += n; return p; };

  float* MODSH = alloc((size_t)NNODE * 1536);
  float* MEANB = alloc(32);
  float* RSTDB = alloc(32);
  float* XNORM = alloc((size_t)NNODE * 256);
  float* XAGG  = alloc((size_t)NNODE * 256);
  float* QKV   = alloc((size_t)NNODE * 768);
  float* QN    = alloc((size_t)NNODE * 256);
  float* KN    = alloc((size_t)NNODE * 256);
  float* ATTNB = alloc((size_t)NNODE * 256);
  float* X1    = alloc((size_t)NNODE * 256);
  float* TMPN  = alloc((size_t)NNODE * 256);
  float* HN    = alloc((size_t)NNODE * 256);
  __bf16* SN   = (__bf16*)alloc((size_t)NNODE * SW / 2);
  float* ENORM = alloc((size_t)NEDGE * 256);
  float* HIDE  = alloc((size_t)NEDGE * 256);   // fe hidden, later m
  float* MSG   = alloc((size_t)NEDGE * 256);   // messages, later EA1
  float* TMPE  = alloc((size_t)NEDGE * 256);
  float* HEB   = alloc((size_t)NEDGE * 256);
  float* EMOD  = alloc((size_t)NEDGE * 512);   // two 256-wide modulation chunks
  __bf16* SE   = (__bf16*)alloc((size_t)NEDGE * SW / 2);
  float* EA1   = MSG;
  float* EMODA = EMOD;
  const int SCOFF_E = NEDGE * 256;

  const dim3 B256(256);
  const dim3 GE2(NEDGE / 64, 2);   // Nout=256, NT=4
  const dim3 GN2(NNODE / 64, 2);

  // mods_h = silu(t_emb_h) @ adaln_w + adaln_b  [N,1536]
  { GArgs g{}; g.A = t_emb_h; g.lda = HIDDEN; g.W = adaln_w; g.ldw = 1536; g.bias = adaln_b;
    g.out = MODSH; g.ldo = 1536; g.M = NNODE; g.N = 1536; g.K = HIDDEN;
    gemm_wmma<A_SILU, EP_BIAS, 4><<<dim3(NNODE / 64, 12), B256, 0, stream>>>(g); }

  // x_norm = modulate(BLN(x,1e-6), shift_msa, scale_msa)
  bln_stats<<<NGRAPH, B256, 0, stream>>>(xin, HIDDEN, 0, NODESPER, 1e-6f, MEANB, RSTDB);
  bln_apply<true, false><<<NNODE, B256, 0, stream>>>(xin, HIDDEN, 0, XNORM, HIDDEN, NODESPER,
      MEANB, RSTDB, MODSH, 1536, 0, 256, nullptr, nullptr, NNODE);

  // edge modulation chunks: shift_msa -> EMOD_A, scale_msa -> EMOD_B
  { GArgs g{}; g.A = t_emb_e; g.lda = HIDDEN; g.ldw = 1536; g.ldo = HIDDEN;
    g.M = NEDGE; g.N = HIDDEN; g.K = HIDDEN;
    g.W = adaln_e_w;       g.bias = adaln_e_b;       g.out = EMODA;
    gemm_wmma<A_SILU, EP_BIAS, 4><<<GE2, B256, 0, stream>>>(g);
    g.W = adaln_e_w + 256; g.bias = adaln_e_b + 256; g.out = EMODA + (size_t)NEDGE * 256;
    gemm_wmma<A_SILU, EP_BIAS, 4><<<GE2, B256, 0, stream>>>(g); }

  // e_norm = modulate(BLN(edge_attr,1e-6), e_shift_msa, e_scale_msa)
  bln_stats<<<NGRAPH, B256, 0, stream>>>(edge_attr, HIDDEN, 0, EPERG, 1e-6f, MEANB, RSTDB);
  bln_apply<true, false><<<NEDGE, B256, 0, stream>>>(edge_attr, HIDDEN, 0, ENORM, HIDDEN, EPERG,
      MEANB, RSTDB, EMOD, 256, 0, SCOFF_E, nullptr, nullptr, NEDGE);

  // hidden = silu(concat(x_norm[src],x_norm[tgt],e_norm,dist) @ fe_w0 + fe_b0)
  { GArgs g{}; g.W = fe_w0; g.ldw = HIDDEN; g.bias = fe_b0; g.out = HIDE; g.ldo = HIDDEN;
    g.M = NEDGE; g.N = HIDDEN; g.K = FEIN;
    g.src = src; g.tgt = tgt; g.xn = XNORM; g.en = ENORM; g.dist = dist;
    gemm_wmma<A_FEATS, EP_BIAS_SILU, 4><<<GE2, B256, 0, stream>>>(g); }

  // messages = hidden @ fe_w1 + fe_b1
  { GArgs g{}; g.A = HIDE; g.lda = HIDDEN; g.W = fe_w1; g.ldw = HIDDEN; g.bias = fe_b1;
    g.out = MSG; g.ldo = HIDDEN; g.M = NEDGE; g.N = HIDDEN; g.K = HIDDEN;
    gemm_wmma<A_PLAIN, EP_BIAS, 4><<<GE2, B256, 0, stream>>>(g); }

  // x_agg = scatter_mean(messages, src) — contiguous 47-edge blocks
  agg_mean<<<NNODE, B256, 0, stream>>>(MSG, XAGG);

  // qkv = x_agg @ qkv_w
  { GArgs g{}; g.A = XAGG; g.lda = HIDDEN; g.W = qkv_w; g.ldw = 768;
    g.out = QKV; g.ldo = 768; g.M = NNODE; g.N = 768; g.K = HIDDEN;
    gemm_wmma<A_PLAIN, EP_NONE, 4><<<dim3(NNODE / 64, 6), B256, 0, stream>>>(g); }

  // Q/K per-graph layernorm
  bln_stats<<<NGRAPH, B256, 0, stream>>>(QKV, 768, 0, NODESPER, 1e-6f, MEANB, RSTDB);
  bln_apply<false, false><<<NNODE, B256, 0, stream>>>(QKV, 768, 0, QN, HIDDEN, NODESPER,
      MEANB, RSTDB, nullptr, 0, 0, 0, nullptr, nullptr, NNODE);
  bln_stats<<<NGRAPH, B256, 0, stream>>>(QKV, 768, 256, NODESPER, 1e-6f, MEANB, RSTDB);
  bln_apply<false, false><<<NNODE, B256, 0, stream>>>(QKV, 768, 256, KN, HIDDEN, NODESPER,
      MEANB, RSTDB, nullptr, 0, 0, 0, nullptr, nullptr, NNODE);

  // block-diagonal attention
  attn_kernel<<<dim3(NGRAPH, HEADS), 64, 0, stream>>>(QN, KN, QKV, ATTNB);

  // x1 = x + gate_msa * (attn_out @ out_w)
  { GArgs g{}; g.A = ATTNB; g.lda = HIDDEN; g.W = out_w; g.ldw = HIDDEN;
    g.out = X1; g.ldo = HIDDEN; g.M = NNODE; g.N = HIDDEN; g.K = HIDDEN;
    g.res = xin; g.ldres = HIDDEN; g.gate = MODSH; g.ldg = 1536; g.goff = 512;
    gemm_wmma<A_PLAIN, EP_GATED, 4><<<GN2, B256, 0, stream>>>(g); }

  // h = BLN(modulate(BLN(x1,1e-6), shift_mlp, scale_mlp), 1e-5, w, b)
  bln_stats<<<NGRAPH, B256, 0, stream>>>(X1, HIDDEN, 0, NODESPER, 1e-6f, MEANB, RSTDB);
  bln_apply<true, false><<<NNODE, B256, 0, stream>>>(X1, HIDDEN, 0, TMPN, HIDDEN, NODESPER,
      MEANB, RSTDB, MODSH, 1536, 768, 1024, nullptr, nullptr, NNODE);
  bln_stats<<<NGRAPH, B256, 0, stream>>>(TMPN, HIDDEN, 0, NODESPER, 1e-5f, MEANB, RSTDB);
  bln_apply<false, true><<<NNODE, B256, 0, stream>>>(TMPN, HIDDEN, 0, HN, HIDDEN, NODESPER,
      MEANB, RSTDB, nullptr, 0, 0, 0, ffn_norm_w, ffn_norm_b, NNODE);

  // node SwiGLU: SN = silu(a)*b (bf16), then x2 = x1 + gate_mlp * (SN @ ffn_w_out)
  gemm_swiglu<<<dim3(NNODE / 64, SW / 128), B256, 0, stream>>>(HN, HIDDEN, ffn_w_in, 2 * SW, SN, SW, NNODE);
  { GArgs g{}; g.A = (const float*)SN; g.lda = SW; g.W = ffn_w_out; g.ldw = HIDDEN;
    g.out = out_x; g.ldo = HIDDEN; g.M = NNODE; g.N = HIDDEN; g.K = SW;
    g.res = X1; g.ldres = HIDDEN; g.gate = MODSH; g.ldg = 1536; g.goff = 1280;
    gemm_wmma<A_BF16, EP_GATED, 4><<<GN2, B256, 0, stream>>>(g); }

  // m = (x2[src] + x2[tgt]) @ lin_e0_w   (reuse HIDE)
  { GArgs g{}; g.A = out_x; g.lda = HIDDEN; g.W = lin_e0_w; g.ldw = HIDDEN;
    g.out = HIDE; g.ldo = HIDDEN; g.M = NEDGE; g.N = HIDDEN; g.K = HIDDEN;
    g.src = src; g.tgt = tgt;
    gemm_wmma<A_SUMPAIR, EP_NONE, 4><<<GE2, B256, 0, stream>>>(g); }

  // e_gate_msa chunk -> EMOD_A
  { GArgs g{}; g.A = t_emb_e; g.lda = HIDDEN; g.W = adaln_e_w + 512; g.ldw = 1536;
    g.bias = adaln_e_b + 512; g.out = EMODA; g.ldo = HIDDEN;
    g.M = NEDGE; g.N = HIDDEN; g.K = HIDDEN;
    gemm_wmma<A_SILU, EP_BIAS, 4><<<GE2, B256, 0, stream>>>(g); }

  // EA1 = edge_attr + e_gate_msa * (concat[m, dist] @ lin_e1_w)
  { GArgs g{}; g.A = HIDE; g.lda = HIDDEN; g.W = lin_e1_w; g.ldw = HIDDEN;
    g.out = EA1; g.ldo = HIDDEN; g.M = NEDGE; g.N = HIDDEN; g.K = HIDDEN + DISTD;
    g.dist = dist; g.res = edge_attr; g.ldres = HIDDEN; g.gate = EMODA; g.ldg = HIDDEN; g.goff = 0;
    gemm_wmma<A_CATMD, EP_GATED, 4><<<GE2, B256, 0, stream>>>(g); }

  // e_shift_mlp -> EMOD_A, e_scale_mlp -> EMOD_B
  { GArgs g{}; g.A = t_emb_e; g.lda = HIDDEN; g.ldw = 1536; g.ldo = HIDDEN;
    g.M = NEDGE; g.N = HIDDEN; g.K = HIDDEN;
    g.W = adaln_e_w + 768;  g.bias = adaln_e_b + 768;  g.out = EMODA;
    gemm_wmma<A_SILU, EP_BIAS, 4><<<GE2, B256, 0, stream>>>(g);
    g.W = adaln_e_w + 1024; g.bias = adaln_e_b + 1024; g.out = EMODA + (size_t)NEDGE * 256;
    gemm_wmma<A_SILU, EP_BIAS, 4><<<GE2, B256, 0, stream>>>(g); }

  // he = BLN(modulate(BLN(EA1,1e-6), e_shift_mlp, e_scale_mlp), 1e-5, w, b)
  bln_stats<<<NGRAPH, B256, 0, stream>>>(EA1, HIDDEN, 0, EPERG, 1e-6f, MEANB, RSTDB);
  bln_apply<true, false><<<NEDGE, B256, 0, stream>>>(EA1, HIDDEN, 0, TMPE, HIDDEN, EPERG,
      MEANB, RSTDB, EMOD, 256, 0, SCOFF_E, nullptr, nullptr, NEDGE);
  bln_stats<<<NGRAPH, B256, 0, stream>>>(TMPE, HIDDEN, 0, EPERG, 1e-5f, MEANB, RSTDB);
  bln_apply<false, true><<<NEDGE, B256, 0, stream>>>(TMPE, HIDDEN, 0, HEB, HIDDEN, EPERG,
      MEANB, RSTDB, nullptr, 0, 0, 0, ffn_norm_e_w, ffn_norm_e_b, NEDGE);

  // edge SwiGLU
  gemm_swiglu<<<dim3(NEDGE / 64, SW / 128), B256, 0, stream>>>(HEB, HIDDEN, ffn_e_w_in, 2 * SW, SE, SW, NEDGE);

  // e_gate_mlp chunk -> EMOD_A
  { GArgs g{}; g.A = t_emb_e; g.lda = HIDDEN; g.W = adaln_e_w + 1280; g.ldw = 1536;
    g.bias = adaln_e_b + 1280; g.out = EMODA; g.ldo = HIDDEN;
    g.M = NEDGE; g.N = HIDDEN; g.K = HIDDEN;
    gemm_wmma<A_SILU, EP_BIAS, 4><<<GE2, B256, 0, stream>>>(g); }

  // edge_attr_out = EA1 + e_gate_mlp * (SE @ ffn_e_w_out)
  { GArgs g{}; g.A = (const float*)SE; g.lda = SW; g.W = ffn_e_w_out; g.ldw = HIDDEN;
    g.out = out_e; g.ldo = HIDDEN; g.M = NEDGE; g.N = HIDDEN; g.K = SW;
    g.res = EA1; g.ldres = HIDDEN; g.gate = EMODA; g.ldg = HIDDEN; g.goff = 0;
    gemm_wmma<A_BF16, EP_GATED, 4><<<GE2, B256, 0, stream>>>(g); }
}
